// GCN_DANN_12214886990280
// MI455X (gfx1250) — matrix operand order repaired and checked by reading, not compiled, not run on hardware
//
#include <hip/hip_runtime.h>
#include <hip/hip_bf16.h>

// ---------------------------------------------------------------------------
// GCN-DANN forward for MI455X (gfx1250, wave32).
//   - node GEMMs on v_wmma_f32_16x16x32_bf16 (f32 accumulate)
//     * W pre-packed into per-lane B-fragment layout -> 2x global b128 / frag
//     * A fragments: 2x ds_load_b128 + shufflevector (contiguous runs)
//     * 64 rows/block, B fragments register-resident across 4 M-tiles
//     * uniform full-tile fast path: immediate-offset stores, no guards
//   - edge aggregation: 1 wave per edge, float4/lane, HW f32 global atomics
//   - L2-resident working set (2 x 51.2MB < 192MB L2)
// ---------------------------------------------------------------------------

#define C_DIM   128
#define HID     64
#define NCLS    10
#define BN_EPS  1e-5f

typedef __attribute__((ext_vector_type(16))) __bf16 v16bf;
typedef __attribute__((ext_vector_type(8)))  __bf16 v8bf;
typedef __attribute__((ext_vector_type(4)))  __bf16 v4bf;
typedef __attribute__((ext_vector_type(8)))  float  v8f;

__device__ __forceinline__ void atomAddF(float* p, float v) {
  unsafeAtomicAdd(p, v);   // lowers to global_atomic_add_f32 on gfx1250
}

// ---------------- degree / norm ----------------
__global__ void k_init_deg(float* deg, int n) {
  int i = blockIdx.x * blockDim.x + threadIdx.x;
  if (i < n) deg[i] = 1.0f;                 // self loop
}

__global__ void k_deg_edges(const int* __restrict__ dst, float* deg, int E) {
  int e = blockIdx.x * blockDim.x + threadIdx.x;
  if (e < E) atomAddF(&deg[dst[e]], 1.0f);
}

__global__ void k_dinv(float* deg, int n) {
  int i = blockIdx.x * blockDim.x + threadIdx.x;
  if (i < n) {
    float d = deg[i];
    deg[i] = (d > 0.0f) ? rsqrtf(d) : 0.0f; // in place -> dinv
  }
}

// ---------------- generic zero ----------------
__global__ void k_zero(float* p, int n) {
  int i = blockIdx.x * blockDim.x + threadIdx.x;
  if (i < n) p[i] = 0.0f;
}

// ---------------- pack W[128x128] f32 -> per-lane B-fragment bf16 layout ---
// Packed index (per layer): (((ct*4 + kc)*32 + lane)*16 + e)
//   source: K = 32*kc + 8*(lane>>4) + (e&7) + 16*(e>>3), Ncol = 16*ct + (lane&15)
__global__ void k_pack_W(const float* __restrict__ Wc, __bf16* __restrict__ Wp,
                         int total) {
  int idx = blockIdx.x * blockDim.x + threadIdx.x;
  if (idx >= total) return;
  int l  = idx >> 14;          // 16384 packed elems per layer
  int r  = idx & 16383;
  int ct = r >> 11;
  int kc = (r >> 9) & 3;
  int lane = (r >> 4) & 31;
  int e  = r & 15;
  int half = lane >> 4, lm = lane & 15;
  int v  = e >> 1;
  int kb = kc * 32 + 8 * half + (e & 7) + 16 * (e >> 3);
  int nn = ct * 16 + lm;
  Wp[idx] = (__bf16)Wc[l * 16384 + kb * C_DIM + nn];
}

// ---------------- GEMM: out[N x 128] = in[N x 128] @ W[128 x 128] ----------
// 256 threads = 8 waves. Block covers 64 rows (4 M-tiles of 16 per wave);
// wave w covers cols [16w,16w+16); B fragments stay in registers.
__global__ void k_gemm_bf16(const float* __restrict__ in,
                            const __bf16* __restrict__ Wp,
                            float* __restrict__ out, int n) {
  __shared__ __bf16 sA[64 * C_DIM];      // 16 KB, [m][k]

  const int  tid     = threadIdx.x;
  const int  rowBase = blockIdx.x * 64;
  const bool full    = (rowBase + 64 <= n);   // uniform per block

  // stage A tile: 2048 float4 (8/thread) -> bf16x4 LDS stores
  if (full) {
    #pragma unroll
    for (int i = 0; i < 8; ++i) {
      int idx4 = tid + 256 * i;              // float4 index in 64x128 tile
      float4 v = ((const float4*)(in + (size_t)rowBase * C_DIM))[idx4];
      v4bf b = {(__bf16)v.x, (__bf16)v.y, (__bf16)v.z, (__bf16)v.w};
      *(v4bf*)&sA[idx4 * 4] = b;
    }
  } else {
    #pragma unroll
    for (int i = 0; i < 8; ++i) {
      int idx4 = tid + 256 * i;
      int r = (idx4 * 4) >> 7;
      float4 v = make_float4(0.f, 0.f, 0.f, 0.f);
      if (rowBase + r < n)
        v = ((const float4*)(in + (size_t)rowBase * C_DIM))[idx4];
      v4bf b = {(__bf16)v.x, (__bf16)v.y, (__bf16)v.z, (__bf16)v.w};
      *(v4bf*)&sA[idx4 * 4] = b;
    }
  }
  __syncthreads();

  const int lane    = tid & 31;
  const int wave    = tid >> 5;
  const int half    = lane >> 4;       // 0: lanes 0-15, 1: lanes 16-31
  const int lm      = lane & 15;
  const int colBase = wave * 16;

  // B fragments for this wave's column tile: 32 contiguous bytes per lane
  v16bf bfrag[4];
  #pragma unroll
  for (int kc = 0; kc < 4; ++kc)
    bfrag[kc] = *(const v16bf*)(Wp + (((size_t)wave * 4 + kc) * 32 + lane) * 16);

  // per-lane output base: row (rowBase + 8*half), col (colBase + lm)
  float* outBase = out + ((size_t)rowBase + 8 * half) * C_DIM + colBase + lm;

  #pragma unroll
  for (int mt = 0; mt < 4; ++mt) {
    v8f acc = {0.f, 0.f, 0.f, 0.f, 0.f, 0.f, 0.f, 0.f};
    #pragma unroll
    for (int kc = 0; kc < 4; ++kc) {
      // A 16x32 bf16 lane layout = two contiguous 8-elem runs
      const __bf16* ap = sA + (mt * 16 + lm) * C_DIM + kc * 32 + 8 * half;
      v8bf alo = *(const v8bf*)ap;          // K = 32kc+8h .. +7
      v8bf ahi = *(const v8bf*)(ap + 16);   // K = 32kc+16+8h .. +7
      v16bf a = __builtin_shufflevector(alo, ahi,
          0, 1, 2, 3, 4, 5, 6, 7, 8, 9, 10, 11, 12, 13, 14, 15);
      acc = __builtin_amdgcn_wmma_f32_16x16x32_bf16(
          false, a, false, bfrag[kc], (short)0, acc, false, false);
    }
    // C/D layout: row m = v + 8*half (consecutive), col = colBase + lm
    float* po = outBase + (size_t)mt * 16 * C_DIM;
    if (full) {
      #pragma unroll
      for (int v = 0; v < 8; ++v) po[(size_t)v * C_DIM] = acc[v];
    } else {
      #pragma unroll
      for (int v = 0; v < 8; ++v) {
        int gr = rowBase + mt * 16 + v + 8 * half;
        if (gr < n) po[(size_t)v * C_DIM] = acc[v];
      }
    }
  }
}

// ---------------- edge aggregation: acc[dst] += hw[src]*dinv[s]*dinv[d] ----
// one wave per edge; lane handles float4 (32*4 = 128 features)
__global__ void k_edge_agg(const int* __restrict__ src,
                           const int* __restrict__ dst,
                           const float* __restrict__ hw,
                           const float* __restrict__ dinv,
                           float* __restrict__ acc, int E) {
  int e = blockIdx.x * 8 + (threadIdx.x >> 5);
  if (e >= E) return;
  int lane = threadIdx.x & 31;
  int s = src[e], d = dst[e];
  float nrm = dinv[s] * dinv[d];
  const float4 v = ((const float4*)(hw + (size_t)s * C_DIM))[lane];
  float* p = acc + (size_t)d * C_DIM + lane * 4;
  atomAddF(p + 0, v.x * nrm);
  atomAddF(p + 1, v.y * nrm);
  atomAddF(p + 2, v.z * nrm);
  atomAddF(p + 3, v.w * nrm);
}

// ---------------- finalize: self-loop + bias + BN(eval) + ReLU, in place ---
// float4-vectorized over features
__global__ void k_finalize(float* __restrict__ acc,
                           const float* __restrict__ hw,
                           const float* __restrict__ dinv,
                           const float* __restrict__ bc,
                           const float* __restrict__ gamma,
                           const float* __restrict__ beta,
                           const float* __restrict__ mean,
                           const float* __restrict__ var, int n) {
  int idx = blockIdx.x * blockDim.x + threadIdx.x;   // over n*32 float4s
  if (idx >= n * (C_DIM / 4)) return;
  int node = idx >> 5, c4 = idx & 31;
  float di = dinv[node];
  float dd = di * di;
  const float4 a  = ((const float4*)acc)[idx];
  const float4 hv = ((const float4*)hw)[idx];
  const float4 b  = ((const float4*)bc)[c4];
  const float4 gm = ((const float4*)gamma)[c4];
  const float4 bt = ((const float4*)beta)[c4];
  const float4 mn = ((const float4*)mean)[c4];
  const float4 vr = ((const float4*)var)[c4];
  float4 r;
  r.x = fmaxf((a.x + hv.x * dd + b.x - mn.x) * rsqrtf(vr.x + BN_EPS) * gm.x + bt.x, 0.f);
  r.y = fmaxf((a.y + hv.y * dd + b.y - mn.y) * rsqrtf(vr.y + BN_EPS) * gm.y + bt.y, 0.f);
  r.z = fmaxf((a.z + hv.z * dd + b.z - mn.z) * rsqrtf(vr.z + BN_EPS) * gm.z + bt.z, 0.f);
  r.w = fmaxf((a.w + hv.w * dd + b.w - mn.w) * rsqrtf(vr.w + BN_EPS) * gm.w + bt.w, 0.f);
  ((float4*)acc)[idx] = r;
}

// ---------------- mean pool ----------------
__global__ void k_pool_accum(const float* __restrict__ h,
                             const int* __restrict__ batch,
                             float* __restrict__ sums,
                             float* __restrict__ cnts, int n) {
  int node = blockIdx.x * 8 + (threadIdx.x >> 5);
  if (node >= n) return;
  int lane = threadIdx.x & 31;
  int g = batch[node];
  const float4 v = ((const float4*)(h + (size_t)node * C_DIM))[lane];
  float* p = sums + (size_t)g * C_DIM + lane * 4;
  atomAddF(p + 0, v.x);
  atomAddF(p + 1, v.y);
  atomAddF(p + 2, v.z);
  atomAddF(p + 3, v.w);
  if (lane == 0) atomAddF(&cnts[g], 1.0f);
}

__global__ void k_pool_div(const float* __restrict__ sums,
                           const float* __restrict__ cnts,
                           float* __restrict__ feat, int G) {
  int idx = blockIdx.x * blockDim.x + threadIdx.x;
  if (idx >= G * C_DIM) return;
  int g = idx >> 7;
  feat[idx] = sums[idx] / fmaxf(cnts[g], 1.0f);
}

// ---------------- heads: class + domain (GRL is identity in fwd) ----------
__global__ void k_heads(const float* __restrict__ feat,
                        const float* __restrict__ Wcls,
                        const float* __restrict__ bcls,
                        const float* __restrict__ Wd1,
                        const float* __restrict__ bd1,
                        const float* __restrict__ Wd2,
                        const float* __restrict__ bd2,
                        float* __restrict__ out_class,
                        float* __restrict__ out_domain, int G) {
  __shared__ float sF[C_DIM];
  __shared__ float sH[HID];
  int g = blockIdx.x;
  int t = threadIdx.x;
  sF[t] = feat[(size_t)g * C_DIM + t];
  __syncthreads();

  if (t < NCLS) {
    float s = bcls[t];
    #pragma unroll 8
    for (int k = 0; k < C_DIM; ++k) s += sF[k] * Wcls[k * NCLS + t];
    out_class[g * NCLS + t] = s;
  }
  if (t < HID) {
    float s = bd1[t];
    #pragma unroll 8
    for (int k = 0; k < C_DIM; ++k) s += sF[k] * Wd1[k * HID + t];
    sH[t] = fmaxf(s, 0.0f);
  }
  __syncthreads();
  if (t < 2) {
    float s = bd2[t];
    #pragma unroll 8
    for (int k = 0; k < HID; ++k) s += sH[k] * Wd2[k * 2 + t];
    out_domain[g * 2 + t] = s;
  }
}

// ---------------------------------------------------------------------------
extern "C" void kernel_launch(void* const* d_in, const int* in_sizes, int n_in,
                              void* d_out, int out_size, void* d_ws, size_t ws_size,
                              hipStream_t stream) {
  const float* x     = (const float*)d_in[0];
  const int*   ei    = (const int*)d_in[1];
  const int*   batch = (const int*)d_in[2];
  const float* Wc    = (const float*)d_in[3];
  const float* bc    = (const float*)d_in[4];
  const float* gamma = (const float*)d_in[5];
  const float* beta  = (const float*)d_in[6];
  const float* rmean = (const float*)d_in[7];
  const float* rvar  = (const float*)d_in[8];
  const float* Wcls  = (const float*)d_in[9];
  const float* bcls  = (const float*)d_in[10];
  const float* Wd1   = (const float*)d_in[11];
  const float* bd1   = (const float*)d_in[12];
  const float* Wd2   = (const float*)d_in[13];
  const float* bd2   = (const float*)d_in[14];

  const int N = in_sizes[0] / C_DIM;
  const int E = in_sizes[1] / 2;
  const int G = out_size / (NCLS + 2 + C_DIM);   // 140 floats per graph

  const int* srcp = ei;       // edge_index[0]
  const int* dstp = ei + E;   // edge_index[1]

  // workspace carve-up (256B aligned)
  char* w = (char*)d_ws;
  auto carve = [&](size_t bytes) {
    void* p = (void*)w;
    w += (bytes + 255) & ~(size_t)255;
    return p;
  };
  float*  dinv = (float*)carve((size_t)N * sizeof(float));
  __bf16* Wp   = (__bf16*)carve((size_t)3 * 16384 * sizeof(__bf16));
  float*  buf1 = (float*)carve((size_t)N * C_DIM * sizeof(float)); // hw
  float*  buf2 = (float*)carve((size_t)N * C_DIM * sizeof(float)); // acc / h
  float*  sums = (float*)carve((size_t)G * C_DIM * sizeof(float));
  float*  cnts = (float*)carve((size_t)G * sizeof(float));

  float* out_class  = (float*)d_out;                  // [G,10]
  float* out_domain = out_class + (size_t)G * NCLS;   // [G,2]
  float* out_feat   = out_domain + (size_t)G * 2;     // [G,128]

  const int T = 256;
  // degree / dinv
  k_init_deg<<<(N + T - 1) / T, T, 0, stream>>>(dinv, N);
  k_deg_edges<<<(E + T - 1) / T, T, 0, stream>>>(dstp, dinv, E);
  k_dinv<<<(N + T - 1) / T, T, 0, stream>>>(dinv, N);

  // pack all 3 layer weights into B-fragment layout
  const int packTotal = 3 * 16384;
  k_pack_W<<<(packTotal + T - 1) / T, T, 0, stream>>>(Wc, Wp, packTotal);

  const int nodeElems = N * C_DIM;
  const int nodeVec4  = N * (C_DIM / 4);
  const int gemmBlocks = (N + 63) / 64;
  const int edgeBlocks = (E + 7) / 8;
  const int nodeWaveBlocks = (N + 7) / 8;

  // 3 GCN layers (ping-pong: buf2 holds h, buf1 holds hw)
  for (int l = 0; l < 3; ++l) {
    const float* hin = (l == 0) ? x : buf2;
    k_gemm_bf16<<<gemmBlocks, T, 0, stream>>>(hin, Wp + (size_t)l * 16384,
                                              buf1, N);
    k_zero<<<(nodeElems + T - 1) / T, T, 0, stream>>>(buf2, nodeElems);
    k_edge_agg<<<edgeBlocks, T, 0, stream>>>(srcp, dstp, buf1, dinv, buf2, E);
    k_finalize<<<(nodeVec4 + T - 1) / T, T, 0, stream>>>(
        buf2, buf1, dinv, bc + l * C_DIM, gamma + l * C_DIM, beta + l * C_DIM,
        rmean + l * C_DIM, rvar + l * C_DIM, N);
  }

  // mean pool
  k_zero<<<(G * C_DIM + T - 1) / T, T, 0, stream>>>(sums, G * C_DIM);
  k_zero<<<(G + T - 1) / T, T, 0, stream>>>(cnts, G);
  k_pool_accum<<<nodeWaveBlocks, T, 0, stream>>>(buf2, batch, sums, cnts, N);
  k_pool_div<<<(G * C_DIM + T - 1) / T, T, 0, stream>>>(sums, cnts, out_feat, G);

  // heads
  k_heads<<<G, C_DIM, 0, stream>>>(out_feat, Wcls, bcls, Wd1, bd1, Wd2, bd2,
                                   out_class, out_domain, G);
}
